// EnhancedGNNModel_89773406421144
// MI455X (gfx1250) — matrix operand order repaired and checked by reading, not compiled, not run on hardware
//
#include <hip/hip_runtime.h>
#include <math.h>

// ---------------------------------------------------------------------------
// Types for WMMA (gfx1250, wave32)
// ---------------------------------------------------------------------------
typedef __bf16          bf16x16 __attribute__((ext_vector_type(16)));
typedef unsigned short  u16x8   __attribute__((ext_vector_type(8)));
typedef unsigned short  u16x16  __attribute__((ext_vector_type(16)));
typedef float           f32x8   __attribute__((ext_vector_type(8)));

__device__ __forceinline__ unsigned short f2bf(float f) {
  unsigned int u = __float_as_uint(f);
  unsigned int r = (u + 0x7fffu + ((u >> 16) & 1u)) >> 16;   // round-nearest-even
  return (unsigned short)r;
}

// order-preserving float<->uint map for atomicMax on signed floats
__device__ __forceinline__ unsigned int fenc(float f) {
  unsigned int u = __float_as_uint(f);
  return (u & 0x80000000u) ? ~u : (u | 0x80000000u);
}
__device__ __forceinline__ float fdec(unsigned int e) {
  return (e & 0x80000000u) ? __uint_as_float(e & 0x7fffffffu)
                           : __uint_as_float(~e);
}
__device__ __forceinline__ float sigmoidf_(float x) { return 1.f / (1.f + expf(-x)); }

// ---------------------------------------------------------------------------
// Tiled GEMM: C[M,N] = relu?(A[M,K] @ B[K,N] + bias)
// A fp32 row-major (row stride lda), B fp32 row-major (row stride ldb==N),
// C fp32 row-major (row stride ldc). fp32 -> bf16 conversion on LDS staging,
// v_wmma_f32_16x16x32_bf16 compute, fp32 accumulate.
// Block: 128 threads = 4 waves. Tile 64x64, K-step 32.
// Wave w computes rows [w*16, w*16+16) x all 64 cols (4 accumulators).
// ---------------------------------------------------------------------------
#define TM 64
#define TN 64
#define TKS 32

__global__ __launch_bounds__(128)
void gemm_bf16_wmma(const float* __restrict__ A, long lda,
                    const float* __restrict__ B, int ldb,
                    const float* __restrict__ bias,
                    float* __restrict__ C, long ldc,
                    int M, int K, int relu)
{
  __shared__ alignas(32) unsigned short As[TM][TKS];   // row-major (m, k)
  __shared__ alignas(32) unsigned short Bs[TN][TKS];   // column-major (n, k)

  const int tid  = threadIdx.x;
  const int wid  = tid >> 5;
  const int lane = tid & 31;
  const long bm  = (long)blockIdx.x * TM;
  const long bn  = (long)blockIdx.y * TN;

  f32x8 acc[4];
#pragma unroll
  for (int c = 0; c < 4; ++c)
#pragma unroll
    for (int j = 0; j < 8; ++j) acc[c][j] = 0.0f;

  const int  rowA  = tid >> 1;          // 0..63
  const int  halfA = (tid & 1) * 16;    // k offset in tile
  const long ga    = bm + rowA;
  const int  colB  = tid >> 1;          // 0..63
  const int  halfB = (tid & 1) * 16;

  // per-lane fragment geometry (wave32; contiguous k halves, same order for A and B)
  const int mfrag = lane & 15;
  const int kbA   = (lane >> 4) * 16;   // A: K half 0..15 or 16..31
  const int kbB   = (lane >> 4) * 16;   // B: K half 0..15 or 16..31

  for (int k0 = 0; k0 < K; k0 += TKS) {
    // stage A tile (guard partial M, zero-pad)
    if (ga < (long)M) {
      const float* ap = A + ga * lda + k0 + halfA;
#pragma unroll
      for (int i = 0; i < 16; i += 4) {
        float4 f = *(const float4*)(ap + i);
        As[rowA][halfA + i + 0] = f2bf(f.x);
        As[rowA][halfA + i + 1] = f2bf(f.y);
        As[rowA][halfA + i + 2] = f2bf(f.z);
        As[rowA][halfA + i + 3] = f2bf(f.w);
      }
      if (k0 + TKS < K) __builtin_prefetch(ap + TKS, 0, 0);
    } else {
#pragma unroll
      for (int i = 0; i < 16; ++i) As[rowA][halfA + i] = 0;
    }
    // stage B tile, transposed into column-major LDS
    {
      const float* bp = B + (long)(k0 + halfB) * ldb + bn + colB;
#pragma unroll
      for (int i = 0; i < 16; ++i)
        Bs[colB][halfB + i] = f2bf(bp[(long)i * ldb]);
    }
    __syncthreads();

    // A fragment: two contiguous 8-element (16B) LDS reads
    u16x8 a0 = *(const u16x8*)&As[wid * 16 + mfrag][kbA];
    u16x8 a1 = *(const u16x8*)&As[wid * 16 + mfrag][kbA + 8];
    u16x16 au = __builtin_shufflevector(a0, a1,
        0, 1, 2, 3, 4, 5, 6, 7, 8, 9, 10, 11, 12, 13, 14, 15);
    bf16x16 afrag = __builtin_bit_cast(bf16x16, au);

#pragma unroll
    for (int c = 0; c < 4; ++c) {
      u16x16 bu = *(const u16x16*)&Bs[c * 16 + (lane & 15)][kbB];
      bf16x16 bfrag = __builtin_bit_cast(bf16x16, bu);
      acc[c] = __builtin_amdgcn_wmma_f32_16x16x32_bf16(
          false, afrag, false, bfrag, (short)0, acc[c], false, false);
    }
    __syncthreads();
  }

  // store: VGPR j -> row j (lanes 0-15) / row j+8 (lanes 16-31), col = lane&15
  const int rbase = wid * 16 + ((lane >> 4) << 3);
  const int cl    = lane & 15;
#pragma unroll
  for (int c = 0; c < 4; ++c) {
    const long gcol = bn + c * 16 + cl;
#pragma unroll
    for (int j = 0; j < 8; ++j) {
      const long grow = bm + rbase + j;
      if (grow < (long)M) {
        float v = acc[c][j];
        if (bias) v += bias[gcol];
        if (relu) v = v > 0.f ? v : 0.f;
        C[grow * ldc + gcol] = v;
      }
    }
  }
}

// ---------------------------------------------------------------------------
// Utility / init kernels
// ---------------------------------------------------------------------------
__global__ void zero_f32_kernel(float* p, long n) {
  long i = (long)blockIdx.x * blockDim.x + threadIdx.x;
  long stride = (long)gridDim.x * blockDim.x;
  for (; i < n; i += stride) p[i] = 0.f;
}

__global__ void init_ms_kernel(unsigned int* mkey, float* s, int n) {
  int i = blockIdx.x * blockDim.x + threadIdx.x;
  if (i < n) { mkey[i] = 0u; s[i] = 0.f; }
}

// ---------------------------------------------------------------------------
// GAT: attention coefficients a_s/a_d per (node, head)
// ---------------------------------------------------------------------------
__global__ void gat_coef_kernel(const float* __restrict__ hbuf,
                                const float* __restrict__ asrc,
                                const float* __restrict__ adst,
                                float* __restrict__ asb, float* __restrict__ adb,
                                int NN)
{
  int idx = blockIdx.x * blockDim.x + threadIdx.x;
  if (idx >= NN * 4) return;
  int n = idx >> 2, hd = idx & 3;
  const float* hr = hbuf + (long)n * 256 + hd * 64;
  float s1 = 0.f, s2 = 0.f;
  for (int d = 0; d < 64; ++d) {
    float hv = hr[d];
    s1 += hv * asrc[hd * 64 + d];
    s2 += hv * adst[hd * 64 + d];
  }
  asb[idx] = s1;
  adb[idx] = s2;
}

// leaky-relu logits + per-(dst,head) segment max via encoded atomicMax
__global__ void gat_logits_kernel(const int* __restrict__ ei,
                                  const float* __restrict__ asb,
                                  const float* __restrict__ adb,
                                  float* __restrict__ lbuf,
                                  unsigned int* __restrict__ mkey,
                                  int E, int NN)
{
  int idx = blockIdx.x * blockDim.x + threadIdx.x;
  int tot = (E + NN) * 4;
  if (idx >= tot) return;
  int e = idx >> 2, hd = idx & 3;
  int src, dst;
  if (e < E) { src = ei[e]; dst = ei[E + e]; } else { src = dst = e - E; }
  float l = asb[src * 4 + hd] + adb[dst * 4 + hd];
  l = l > 0.f ? l : 0.2f * l;
  lbuf[idx] = l;
  atomicMax(&mkey[dst * 4 + hd], fenc(l));
}

__global__ void gat_expsum_kernel(const int* __restrict__ ei,
                                  float* __restrict__ lbuf,
                                  const unsigned int* __restrict__ mkey,
                                  float* __restrict__ sbuf,
                                  int E, int NN)
{
  int idx = blockIdx.x * blockDim.x + threadIdx.x;
  int tot = (E + NN) * 4;
  if (idx >= tot) return;
  int e = idx >> 2, hd = idx & 3;
  int dst = (e < E) ? ei[E + e] : (e - E);
  float m = fdec(mkey[dst * 4 + hd]);
  float w = expf(lbuf[idx] - m);
  lbuf[idx] = w;                       // reuse buffer for exp weights
  atomicAdd(&sbuf[dst * 4 + hd], w);
}

__global__ void gat_scatter_kernel(const int* __restrict__ ei,
                                   const float* __restrict__ hbuf,
                                   const float* __restrict__ lbuf,
                                   const float* __restrict__ sbuf,
                                   float* __restrict__ comb, int colOff,
                                   int E, int NN)
{
  int idx = blockIdx.x * blockDim.x + threadIdx.x;
  int tot = (E + NN) * 4;
  if (idx >= tot) return;
  int e = idx >> 2, hd = idx & 3;
  int src, dst;
  if (e < E) { src = ei[e]; dst = ei[E + e]; } else { src = dst = e - E; }
  float alpha = lbuf[idx] / sbuf[dst * 4 + hd];
  const float* hr = hbuf + (long)src * 256 + hd * 64;
  float* cr = comb + (long)dst * 768 + colOff + hd * 64;
  for (int d = 0; d < 64; ++d)
    atomicAdd(&cr[d], hr[d] * alpha);
}

__global__ void gat_finish_kernel(float* __restrict__ comb,
                                  const float* __restrict__ bias,
                                  int colOff, int NN)
{
  int idx = blockIdx.x * blockDim.x + threadIdx.x;
  if (idx >= NN * 256) return;
  int n = idx >> 8, c = idx & 255;
  float* p = comb + (long)n * 768 + colOff + c;
  float v = *p + bias[c];
  *p = v > 0.f ? v : 0.f;
}

// ---------------------------------------------------------------------------
// SRL attention: per node, 3x3 scores over heads, softmax, o = attn @ v.
// One wave (32 lanes) per node; writes attn straight to d_out.
// ---------------------------------------------------------------------------
__global__ __launch_bounds__(32)
void srl_attn_kernel(const float* __restrict__ q, const float* __restrict__ k,
                     const float* __restrict__ v, float* __restrict__ o,
                     float* __restrict__ attnOut)
{
  const int n = blockIdx.x;
  const int lane = threadIdx.x;
  const float* qr = q + (long)n * 768;
  const float* kr = k + (long)n * 768;
  const float* vr = v + (long)n * 768;
  float sc[9];
#pragma unroll
  for (int p = 0; p < 9; ++p) {
    const int h = p / 3, g = p % 3;
    float part = 0.f;
    for (int d = lane; d < 256; d += 32)
      part += qr[h * 256 + d] * kr[g * 256 + d];
#pragma unroll
    for (int off = 16; off > 0; off >>= 1)
      part += __shfl_xor(part, off);
    sc[p] = part * 0.0625f;            // / sqrt(256)
  }
  float attn[9];
#pragma unroll
  for (int h = 0; h < 3; ++h) {
    float m = fmaxf(sc[h * 3], fmaxf(sc[h * 3 + 1], sc[h * 3 + 2]));
    float e0 = expf(sc[h * 3] - m), e1 = expf(sc[h * 3 + 1] - m),
          e2 = expf(sc[h * 3 + 2] - m);
    float s = e0 + e1 + e2;
    attn[h * 3] = e0 / s; attn[h * 3 + 1] = e1 / s; attn[h * 3 + 2] = e2 / s;
  }
  if (lane < 9) attnOut[(long)n * 9 + lane] = attn[lane];
  for (int idx = lane; idx < 768; idx += 32) {
    int h = idx >> 8, d = idx & 255;
    o[(long)n * 768 + idx] = attn[h * 3 + 0] * vr[d] +
                             attn[h * 3 + 1] * vr[256 + d] +
                             attn[h * 3 + 2] * vr[512 + d];
  }
}

// ---------------------------------------------------------------------------
// LSTM scan: Zx precomputed = attended@Wih + bih. Serial over 20000 steps;
// 256 threads, thread j owns gate pre-activation j. Writes h into
// final[:,768:832] (row stride 832).
// ---------------------------------------------------------------------------
__global__ __launch_bounds__(256)
void lstm_kernel(const float* __restrict__ Zx, const float* __restrict__ Whh,
                 const float* __restrict__ bhh, float* __restrict__ fin, int NN)
{
  __shared__ float hs[64], cs[64], zb[256];
  const int j = threadIdx.x;
  if (j < 64) { hs[j] = 0.f; cs[j] = 0.f; }
  __syncthreads();
  const float bj = bhh[j];
  for (int t = 0; t < NN; ++t) {
    float z = Zx[(long)t * 256 + j] + bj;
    for (int k2 = 0; k2 < 64; ++k2)
      z += hs[k2] * Whh[k2 * 256 + j];
    zb[j] = z;
    __syncthreads();
    if (j < 64) {
      float ig = sigmoidf_(zb[j]);
      float fg = sigmoidf_(zb[64 + j]);
      float gg = tanhf(zb[128 + j]);
      float og = sigmoidf_(zb[192 + j]);
      float c = fg * cs[j] + ig * gg;
      float h = og * tanhf(c);
      cs[j] = c; hs[j] = h;
      fin[(long)t * 832 + 768 + j] = h;
    }
    __syncthreads();
  }
}

// ---------------------------------------------------------------------------
// Output heads: scores (64->1) and recs (64->5) from relu'd hidden buffers
// ---------------------------------------------------------------------------
__global__ void head_out_kernel(const float* __restrict__ sh,
                                const float* __restrict__ rh,
                                const float* __restrict__ Ws2, const float* __restrict__ bs2,
                                const float* __restrict__ Wr2, const float* __restrict__ br2,
                                float* __restrict__ out, int NN)
{
  int n = blockIdx.x * blockDim.x + threadIdx.x;
  if (n >= NN) return;
  float a = bs2[0];
  for (int k = 0; k < 64; ++k) a += sh[(long)n * 64 + k] * Ws2[k];
  out[n] = a;
  float* rp = out + NN + (long)n * 5;
  for (int r = 0; r < 5; ++r) {
    float a2 = br2[r];
    for (int k = 0; k < 64; ++k) a2 += rh[(long)n * 64 + k] * Wr2[k * 5 + r];
    rp[r] = a2;
  }
}

__global__ void taskw_kernel(const float* __restrict__ tw, float* __restrict__ out)
{
  if (threadIdx.x == 0 && blockIdx.x == 0) {
    float m = fmaxf(tw[0], tw[1]);
    float e0 = expf(tw[0] - m), e1 = expf(tw[1] - m);
    float s = e0 + e1;
    out[0] = e0 / s;
    out[1] = e1 / s;
  }
}

// ---------------------------------------------------------------------------
// Host-side orchestration
// ---------------------------------------------------------------------------
extern "C" void kernel_launch(void* const* d_in, const int* in_sizes, int n_in,
                              void* d_out, int out_size, void* d_ws, size_t ws_size,
                              hipStream_t stream)
{
  (void)n_in; (void)out_size; (void)ws_size;

  const float* x  = (const float*)d_in[0];
  const int*   ei = (const int*)d_in[1];
  // d_in[2] = edge_attr : unused by the reference
  const float* Wg[3]   = {(const float*)d_in[3],  (const float*)d_in[7],  (const float*)d_in[11]};
  const float* Asrc[3] = {(const float*)d_in[4],  (const float*)d_in[8],  (const float*)d_in[12]};
  const float* Adst[3] = {(const float*)d_in[5],  (const float*)d_in[9],  (const float*)d_in[13]};
  const float* Bg[3]   = {(const float*)d_in[6],  (const float*)d_in[10], (const float*)d_in[14]};
  const float* Wq = (const float*)d_in[15], *bq = (const float*)d_in[16];
  const float* Wk = (const float*)d_in[17], *bk = (const float*)d_in[18];
  const float* Wv = (const float*)d_in[19], *bv = (const float*)d_in[20];
  const float* Wo = (const float*)d_in[21], *bo = (const float*)d_in[22];
  const float* Wih = (const float*)d_in[23], *Whh = (const float*)d_in[24];
  const float* bih = (const float*)d_in[25], *bhh = (const float*)d_in[26];
  const float* Ws1 = (const float*)d_in[27], *bs1 = (const float*)d_in[28];
  const float* Ws2 = (const float*)d_in[29], *bs2 = (const float*)d_in[30];
  const float* Wr1 = (const float*)d_in[31], *br1 = (const float*)d_in[32];
  const float* Wr2 = (const float*)d_in[33], *br2 = (const float*)d_in[34];
  const float* tw  = (const float*)d_in[35];

  const int NN = in_sizes[0] / 256;       // 20000
  const int E  = in_sizes[1] / 2;         // 320000
  const int ET = E + NN;

  float* out = (float*)d_out;
  float* ws  = (float*)d_ws;

  // workspace arena (floats), with lifetime-based aliasing
  const long SZNK = (long)NN * 768;
  float* comb = ws;                 // [NN,768]; later reused as SRL output 'o'
  float* qb   = ws + SZNK;          // [NN,768]; later reused as Zx [NN,256]
  float* kb2  = ws + 2 * SZNK;      // [NN,768]; later reused as head hiddens
  float* vb   = ws + 3 * SZNK;      // [NN,768]; early: GAT scratch
  float* fin  = ws + 4 * SZNK;      // [NN,832]
  // aliases inside vb (all GAT work finishes before v is produced)
  float*        hbuf = vb;                                  // [NN,256]
  float*        lbuf = vb + (long)NN * 256;                 // [ET,4]
  unsigned int* mkey = (unsigned int*)(vb + (long)NN * 256 + (long)ET * 4);
  float*        sbuf = (float*)(mkey + (long)NN * 4);       // [NN,4]
  float*        asb  = sbuf + (long)NN * 4;                 // [NN,4]
  float*        adb  = asb + (long)NN * 4;                  // [NN,4]
  // aliases after SRL consumes q/k/v
  float* zx  = qb;                  // [NN,256]
  float* shb = kb2;                 // [NN,64]
  float* rhb = kb2 + (long)NN * 64; // [NN,64]
  float* obuf = comb;               // SRL output overwrites comb

  float* attnOut = out + (long)NN * 6 + 2;   // after scores [NN] + recs [NN*5] + tw [2]

  const int BLK = 256;
  const dim3 gblk(128);
  auto cdiv = [](long a, long b) { return (int)((a + b - 1) / b); };
  const int MT = cdiv(NN, 64);      // 313 row tiles

  // zero comb (scatter accumulator)
  zero_f32_kernel<<<4096, BLK, 0, stream>>>(comb, SZNK);

  // ----- three GAT branches -----
  for (int b = 0; b < 3; ++b) {
    const int colOff = b * 256;
    // h = x @ W   (M=NN, K=256, N=256)
    gemm_bf16_wmma<<<dim3(MT, 4), gblk, 0, stream>>>(
        x, 256, Wg[b], 256, nullptr, hbuf, 256, NN, 256, 0);
    init_ms_kernel<<<cdiv((long)NN * 4, BLK), BLK, 0, stream>>>(mkey, sbuf, NN * 4);
    gat_coef_kernel<<<cdiv((long)NN * 4, BLK), BLK, 0, stream>>>(
        hbuf, Asrc[b], Adst[b], asb, adb, NN);
    gat_logits_kernel<<<cdiv((long)ET * 4, BLK), BLK, 0, stream>>>(
        ei, asb, adb, lbuf, mkey, E, NN);
    gat_expsum_kernel<<<cdiv((long)ET * 4, BLK), BLK, 0, stream>>>(
        ei, lbuf, mkey, sbuf, E, NN);
    gat_scatter_kernel<<<cdiv((long)ET * 4, BLK), BLK, 0, stream>>>(
        ei, hbuf, lbuf, sbuf, comb, colOff, E, NN);
    gat_finish_kernel<<<cdiv((long)NN * 256, BLK), BLK, 0, stream>>>(
        comb, Bg[b], colOff, NN);
  }

  // ----- SRL projections: q,k,v = comb @ W + b  (M=NN, K=768, N=768) -----
  gemm_bf16_wmma<<<dim3(MT, 12), gblk, 0, stream>>>(comb, 768, Wq, 768, bq, qb,  768, NN, 768, 0);
  gemm_bf16_wmma<<<dim3(MT, 12), gblk, 0, stream>>>(comb, 768, Wk, 768, bk, kb2, 768, NN, 768, 0);
  gemm_bf16_wmma<<<dim3(MT, 12), gblk, 0, stream>>>(comb, 768, Wv, 768, bv, vb,  768, NN, 768, 0);

  // per-node 3x3 attention + o (writes attn to d_out, o into comb region)
  srl_attn_kernel<<<NN, 32, 0, stream>>>(qb, kb2, vb, obuf, attnOut);

  // attended = o @ Wo + bo, written into final[:, :768] (ldc=832)
  gemm_bf16_wmma<<<dim3(MT, 12), gblk, 0, stream>>>(obuf, 768, Wo, 768, bo, fin, 832, NN, 768, 0);

  // LSTM input pre-activation: Zx = attended @ Wih + bih  (K=768, N=256)
  gemm_bf16_wmma<<<dim3(MT, 4), gblk, 0, stream>>>(fin, 832, Wih, 256, bih, zx, 256, NN, 768, 0);

  // serial LSTM scan -> final[:, 768:832]
  lstm_kernel<<<1, 256, 0, stream>>>(zx, Whh, bhh, fin, NN);

  // heads: relu(final @ Ws1 + bs1), relu(final @ Wr1 + br1)  (K=832, N=64)
  gemm_bf16_wmma<<<dim3(MT, 1), gblk, 0, stream>>>(fin, 832, Ws1, 64, bs1, shb, 64, NN, 832, 1);
  gemm_bf16_wmma<<<dim3(MT, 1), gblk, 0, stream>>>(fin, 832, Wr1, 64, br1, rhb, 64, NN, 832, 1);

  head_out_kernel<<<cdiv(NN, BLK), BLK, 0, stream>>>(shb, rhb, Ws2, bs2, Wr2, br2, out, NN);
  taskw_kernel<<<1, 32, 0, stream>>>(tw, out + (long)NN * 6);
}
